// RAttention_22883585753465
// MI455X (gfx1250) — matrix operand order repaired and checked by reading, not compiled, not run on hardware
//
#include <hip/hip_runtime.h>

#define DEVI __device__ __forceinline__

typedef __attribute__((ext_vector_type(16))) __bf16 v16bf;
typedef __attribute__((ext_vector_type(8)))  float  v8f;
typedef __attribute__((ext_vector_type(4)))  unsigned int u32x4;
typedef __attribute__((ext_vector_type(4)))  int    i32x4;
typedef __attribute__((ext_vector_type(8)))  int    i32x8;

constexpr int B_ = 8, C_ = 256, CQ_ = 32, H_ = 128, W_ = 128;
constexpr int HW_ = H_ * W_;

// ---- WMMA fragment index maps (CDNA5 ISA 7.12.2, 16-bit ops, wave32) ----
// A (16x32, MxK): lane = 16*half + m ; element e of v16bf covers K below.
DEVI int kmapA(int e, int half) {
  return ((e >> 3) << 4) + (half << 3) + (((e >> 1) & 3) << 1) + (e & 1);
}
// B (32x16, KxN): lane = 16*g + n ; element e -> K = 16*(e/8) + 8*g + e%8.
DEVI int kmapB(int e, int g) { return ((e >> 3) << 4) + (g << 3) + (e & 7); }
// C/D (16x16 f32): VGPR r, lane l: M = r + 8*(l>>4), N = l&15.

DEVI float rowmax16(float v) {
  v = fmaxf(v, __shfl_xor(v, 8, 16));
  v = fmaxf(v, __shfl_xor(v, 4, 16));
  v = fmaxf(v, __shfl_xor(v, 2, 16));
  v = fmaxf(v, __shfl_xor(v, 1, 16));
  return v;
}
DEVI float rowsum16(float v) {
  v += __shfl_xor(v, 8, 16);
  v += __shfl_xor(v, 4, 16);
  v += __shfl_xor(v, 2, 16);
  v += __shfl_xor(v, 1, 16);
  return v;
}

DEVI v8f wmma_bf16(v16bf a, v16bf b, v8f c) {
  return __builtin_amdgcn_wmma_f32_16x16x32_bf16(false, a, false, b,
                                                 (short)0, c, false, false);
}

// ---- Tensor Data Mover: load a 32(rows) x 128(cols) f32 tile of a 2D tensor
//      (row stride HW_ elements, tensor 256 x 16384) into LDS.  D# per
//      CDNA5 ISA.
DEVI void tdm_load_tile(unsigned lds_off, const float* gaddr) {
  const unsigned long long ga = (unsigned long long)(uintptr_t)gaddr;
  u32x4 g0;
  g0[0] = 1u;                                   // count=1 (valid user D#)
  g0[1] = lds_off;                              // lds_addr (bytes)
  g0[2] = (unsigned)(ga & 0xffffffffu);         // global_addr[31:0]
  g0[3] = (unsigned)(ga >> 32) | 0x80000000u;   // global_addr[56:32] | type=2
  i32x8 g1;
  g1[0] = 0x20000;      // data_size = 2 (4 bytes); no multicast/pad/iterate
  g1[1] = 0x40000000;   // tensor_dim0 = 16384 (low16 at bits 63:48)
  g1[2] = 0x01000000;   // tensor_dim1 = 256   (low16 at bits 95:80)
  g1[3] = 0x00800000;   // tile_dim0  = 128    (bits 127:112)
  g1[4] = 32;           // tile_dim1  = 32     (bits 143:128)
  g1[5] = HW_;          // tensor_dim0_stride[31:0] = 16384
  g1[6] = 0;
  g1[7] = 0;
  const i32x4 z4 = {0, 0, 0, 0};
#if __clang_major__ >= 23
  const i32x8 z8 = {0, 0, 0, 0, 0, 0, 0, 0};
  __builtin_amdgcn_tensor_load_to_lds(g0, g1, z4, z4, z8, 0);
#else
  __builtin_amdgcn_tensor_load_to_lds(g0, g1, z4, z4, 0);
#endif
}

// ============================================================================
// 1) Projection GEMM: out[b,o,p] = sum_c Wt[o,c]*x[b,c,p] + bias[o]  (bf16 out)
//    One block = 16(o) x 128(p) tile, 8 waves. The x panel is staged into LDS
//    by the Tensor Data Mover, double-buffered one K-step ahead (TENSORcnt).
// ============================================================================
__global__ __launch_bounds__(256) void proj_kernel(
    const float* __restrict__ x, const float* __restrict__ Wt,
    const float* __restrict__ bias, __bf16* __restrict__ out, int O) {
  __shared__ float sX[2][32 * 128];  // 2 x 16 KB double buffer
  const int lane = threadIdx.x & 31, wave = threadIdx.x >> 5;
  const int half = lane >> 4, nl = lane & 15;
  const int b = blockIdx.x;
  const int oT = blockIdx.y << 4;
  const int pBase = blockIdx.z << 7;
  const int pl = (wave << 4) + nl;          // column inside the 128-wide tile
  const float* xb = x + (size_t)b * C_ * HW_ + pBase;
  const int orow = oT + nl;

  if (wave == 0)
    tdm_load_tile((unsigned)(uintptr_t)&sX[0][0], xb);  // K-step 0

  v8f acc = {0.f, 0.f, 0.f, 0.f, 0.f, 0.f, 0.f, 0.f};
#pragma unroll 1
  for (int ks = 0; ks < 8; ++ks) {
    if (wave == 0) {
      if (ks < 7) {
        tdm_load_tile((unsigned)(uintptr_t)&sX[(ks + 1) & 1][0],
                      xb + (size_t)(ks + 1) * 32 * HW_);
        __builtin_amdgcn_s_wait_tensorcnt(1);  // K-step ks has landed
      } else {
        __builtin_amdgcn_s_wait_tensorcnt(0);
      }
    }
    __syncthreads();
    const float* sb = sX[ks & 1];
    const int kt = ks << 5;
    v16bf af, bf;
#pragma unroll
    for (int e = 0; e < 16; ++e) {
      af[e] = (__bf16)Wt[orow * C_ + kt + kmapA(e, half)];
      bf[e] = (__bf16)sb[kmapB(e, half) * 128 + pl];
    }
    acc = wmma_bf16(af, bf, acc);
    __syncthreads();  // buffer (ks&1) free for reuse at step ks+2
  }
#pragma unroll
  for (int r = 0; r < 8; ++r) {
    const int o = oT + r + (half << 3);
    out[((size_t)b * O + o) * HW_ + pBase + pl] = (__bf16)(acc[r] + bias[o]);
  }
}

// ============================================================================
// 2) Column stats: per (b,w) compute eH = Qw^T Kw (diag masked),
//    online (max,sumexp) per row h -> mArr/sArr. One WMMA covers K=CQ=32.
// ============================================================================
__global__ __launch_bounds__(256) void colstats_kernel(
    const __bf16* __restrict__ qb, const __bf16* __restrict__ kb,
    float* __restrict__ mArr, float* __restrict__ sArr) {
  const int lane = threadIdx.x & 31, wave = threadIdx.x >> 5;
  const int half = lane >> 4, nl = lane & 15;
  const int b = blockIdx.x, w = blockIdx.y;
  const int hT = wave << 4;
  const size_t qbase = (size_t)b * CQ_ * HW_;
  const float NEG_INF = -__builtin_inff();

  v16bf af;
#pragma unroll
  for (int e = 0; e < 16; ++e)
    af[e] = qb[qbase + (size_t)kmapA(e, half) * HW_ + (hT + nl) * W_ + w];

  float rm[8], rs[8];
#pragma unroll
  for (int r = 0; r < 8; ++r) { rm[r] = NEG_INF; rs[r] = 0.f; }

  for (int jT = 0; jT < 8; ++jT) {
    const int j = (jT << 4) + nl;
    v16bf bf;
#pragma unroll
    for (int e = 0; e < 16; ++e)
      bf[e] = kb[qbase + (size_t)kmapB(e, half) * HW_ + j * W_ + w];
    v8f ee = {0.f, 0.f, 0.f, 0.f, 0.f, 0.f, 0.f, 0.f};
    ee = wmma_bf16(af, bf, ee);
#pragma unroll
    for (int r = 0; r < 8; ++r) {
      const int h = hT + r + (half << 3);
      float v = ee[r];
      if (h == j) v = NEG_INF;                 // diagonal mask
      const float tm = rowmax16(v);
      const float ts = rowsum16(__expf(v - tm));
      const float nm = fmaxf(rm[r], tm);
      rs[r] = rs[r] * __expf(rm[r] - nm) + ts * __expf(tm - nm);
      rm[r] = nm;
    }
  }
  if (nl == 0) {
#pragma unroll
    for (int r = 0; r < 8; ++r) {
      const int h = hT + r + (half << 3);
      const size_t idx = ((size_t)b * H_ + h) * W_ + w;
      mArr[idx] = rm[r]; sArr[idx] = rs[r];
    }
  }
}

// ============================================================================
// 3) Row stats: per (b,h) compute eW = Qh^T Kh, merge (m,s) with column stats.
// ============================================================================
__global__ __launch_bounds__(256) void rowstats_kernel(
    const __bf16* __restrict__ qb, const __bf16* __restrict__ kb,
    float* __restrict__ mArr, float* __restrict__ sArr) {
  const int lane = threadIdx.x & 31, wave = threadIdx.x >> 5;
  const int half = lane >> 4, nl = lane & 15;
  const int b = blockIdx.x, h = blockIdx.y;
  const int wT = wave << 4;
  const size_t qbase = (size_t)b * CQ_ * HW_;
  const float NEG_INF = -__builtin_inff();

  v16bf af;
#pragma unroll
  for (int e = 0; e < 16; ++e)
    af[e] = qb[qbase + (size_t)kmapA(e, half) * HW_ + h * W_ + (wT + nl)];

  float rm[8], rs[8];
#pragma unroll
  for (int r = 0; r < 8; ++r) { rm[r] = NEG_INF; rs[r] = 0.f; }

  for (int jT = 0; jT < 8; ++jT) {
    const int j = (jT << 4) + nl;
    v16bf bf;
#pragma unroll
    for (int e = 0; e < 16; ++e)
      bf[e] = kb[qbase + (size_t)kmapB(e, half) * HW_ + h * W_ + j];
    v8f ee = {0.f, 0.f, 0.f, 0.f, 0.f, 0.f, 0.f, 0.f};
    ee = wmma_bf16(af, bf, ee);
#pragma unroll
    for (int r = 0; r < 8; ++r) {
      const float v = ee[r];
      const float tm = rowmax16(v);
      const float ts = rowsum16(__expf(v - tm));
      const float nm = fmaxf(rm[r], tm);
      rs[r] = rs[r] * __expf(rm[r] - nm) + ts * __expf(tm - nm);
      rm[r] = nm;
    }
  }
  if (nl == 0) {
#pragma unroll
    for (int r = 0; r < 8; ++r) {
      const int wr = wT + r + (half << 3);
      const size_t idx = ((size_t)b * H_ + h) * W_ + wr;
      const float m0 = mArr[idx], s0 = sArr[idx];
      const float nm = fmaxf(m0, rm[r]);
      sArr[idx] = s0 * __expf(m0 - nm) + rs[r] * __expf(rm[r] - nm);
      mArr[idx] = nm;
    }
  }
}

// ============================================================================
// 4) Column output: per (b,w) recompute aH = exp(eH - m)/s into LDS (bf16,
//    transposed), then outH[c,h] = Vw(256x128) x aH^T(128x128). Writes d_out.
// ============================================================================
__global__ __launch_bounds__(256) void colout_kernel(
    const __bf16* __restrict__ qb, const __bf16* __restrict__ kb,
    const __bf16* __restrict__ vb, const float* __restrict__ mArr,
    const float* __restrict__ sArr, float* __restrict__ out) {
  __shared__ __bf16 sAT[HW_];  // aT[j][h], 32 KB
  const int lane = threadIdx.x & 31, wave = threadIdx.x >> 5;
  const int half = lane >> 4, nl = lane & 15;
  const int b = blockIdx.x, w = blockIdx.y;
  const size_t qbase = (size_t)b * CQ_ * HW_;
  const size_t vbase = (size_t)b * C_ * HW_;
  const float NEG_INF = -__builtin_inff();

  // ---- phase 1: attention probabilities into LDS ----
  const int hT = wave << 4;
  v16bf af;
#pragma unroll
  for (int e = 0; e < 16; ++e)
    af[e] = qb[qbase + (size_t)kmapA(e, half) * HW_ + (hT + nl) * W_ + w];
  float mrow[8], sinv[8];
#pragma unroll
  for (int r = 0; r < 8; ++r) {
    const size_t idx = ((size_t)b * H_ + hT + r + (half << 3)) * W_ + w;
    mrow[r] = mArr[idx];
    sinv[r] = 1.0f / sArr[idx];
  }
  for (int jT = 0; jT < 8; ++jT) {
    const int j = (jT << 4) + nl;
    v16bf bf;
#pragma unroll
    for (int e = 0; e < 16; ++e)
      bf[e] = kb[qbase + (size_t)kmapB(e, half) * HW_ + j * W_ + w];
    v8f ee = {0.f, 0.f, 0.f, 0.f, 0.f, 0.f, 0.f, 0.f};
    ee = wmma_bf16(af, bf, ee);
#pragma unroll
    for (int r = 0; r < 8; ++r) {
      const int h = hT + r + (half << 3);
      float v = ee[r];
      if (h == j) v = NEG_INF;
      sAT[j * H_ + h] = (__bf16)(__expf(v - mrow[r]) * sinv[r]);
    }
  }
  __syncthreads();

  // ---- phase 2: outH = V x aH^T ----
  for (int t = wave; t < 128; t += 8) {
    const int cT  = (t >> 3) << 4;
    const int hT2 = (t & 7) << 4;
    v8f acc = {0.f, 0.f, 0.f, 0.f, 0.f, 0.f, 0.f, 0.f};
    for (int jB = 0; jB < H_; jB += 32) {
      v16bf av, bf;
#pragma unroll
      for (int e = 0; e < 16; ++e) {
        av[e] = vb[vbase + (size_t)(cT + nl) * HW_ +
                   (size_t)(jB + kmapA(e, half)) * W_ + w];
        bf[e] = sAT[(jB + kmapB(e, half)) * H_ + hT2 + nl];
      }
      acc = wmma_bf16(av, bf, acc);
    }
#pragma unroll
    for (int r = 0; r < 8; ++r) {
      const int c = cT + r + (half << 3);
      out[vbase + (size_t)c * HW_ + (hT2 + nl) * W_ + w] = acc[r];
    }
  }
}

// ============================================================================
// 5) Row output + epilogue: per (b,h) aW into LDS, outW = V x aW^T, then
//    d_out = gamma*(outH + outW + 2) + x   (outLR+outRL are exactly 1 each).
// ============================================================================
__global__ __launch_bounds__(256) void rowout_kernel(
    const __bf16* __restrict__ qb, const __bf16* __restrict__ kb,
    const __bf16* __restrict__ vb, const float* __restrict__ mArr,
    const float* __restrict__ sArr, const float* __restrict__ x,
    const float* __restrict__ gamma, float* __restrict__ out) {
  __shared__ __bf16 sAT[HW_];  // aT[j][w], 32 KB
  const int lane = threadIdx.x & 31, wave = threadIdx.x >> 5;
  const int half = lane >> 4, nl = lane & 15;
  const int b = blockIdx.x, h = blockIdx.y;
  const size_t qbase = (size_t)b * CQ_ * HW_;
  const size_t vbase = (size_t)b * C_ * HW_;

  // ---- phase 1 ----
  const int wT = wave << 4;
  v16bf af;
#pragma unroll
  for (int e = 0; e < 16; ++e)
    af[e] = qb[qbase + (size_t)kmapA(e, half) * HW_ + h * W_ + (wT + nl)];
  float mrow[8], sinv[8];
#pragma unroll
  for (int r = 0; r < 8; ++r) {
    const size_t idx = ((size_t)b * H_ + h) * W_ + wT + r + (half << 3);
    mrow[r] = mArr[idx];
    sinv[r] = 1.0f / sArr[idx];
  }
  for (int jT = 0; jT < 8; ++jT) {
    const int j = (jT << 4) + nl;
    v16bf bf;
#pragma unroll
    for (int e = 0; e < 16; ++e)
      bf[e] = kb[qbase + (size_t)kmapB(e, half) * HW_ + h * W_ + j];
    v8f ee = {0.f, 0.f, 0.f, 0.f, 0.f, 0.f, 0.f, 0.f};
    ee = wmma_bf16(af, bf, ee);
#pragma unroll
    for (int r = 0; r < 8; ++r) {
      const int wr = wT + r + (half << 3);
      sAT[j * W_ + wr] = (__bf16)(__expf(ee[r] - mrow[r]) * sinv[r]);
    }
  }
  __syncthreads();

  // ---- phase 2 + epilogue ----
  const float g0 = gamma[0];
  for (int t = wave; t < 128; t += 8) {
    const int cT  = (t >> 3) << 4;
    const int wT2 = (t & 7) << 4;
    v8f acc = {0.f, 0.f, 0.f, 0.f, 0.f, 0.f, 0.f, 0.f};
    for (int jB = 0; jB < W_; jB += 32) {
      v16bf av, bf;
#pragma unroll
      for (int e = 0; e < 16; ++e) {
        av[e] = vb[vbase + (size_t)(cT + nl) * HW_ + h * W_ +
                   (jB + kmapA(e, half))];
        bf[e] = sAT[(jB + kmapB(e, half)) * W_ + wT2 + nl];
      }
      acc = wmma_bf16(av, bf, acc);
    }
#pragma unroll
    for (int r = 0; r < 8; ++r) {
      const int c = cT + r + (half << 3);
      const size_t idx = vbase + (size_t)c * HW_ + h * W_ + (wT2 + nl);
      out[idx] = g0 * (out[idx] + acc[r] + 2.0f) + x[idx];
    }
  }
}

// ============================================================================
extern "C" void kernel_launch(void* const* d_in, const int* in_sizes, int n_in,
                              void* d_out, int out_size, void* d_ws,
                              size_t ws_size, hipStream_t stream) {
  (void)in_sizes; (void)n_in; (void)out_size; (void)ws_size;
  const float* x     = (const float*)d_in[0];
  const float* Wq    = (const float*)d_in[1];
  const float* bq    = (const float*)d_in[2];
  const float* Wk    = (const float*)d_in[3];
  const float* bk    = (const float*)d_in[4];
  const float* Wv    = (const float*)d_in[5];
  const float* bv    = (const float*)d_in[6];
  const float* gamma = (const float*)d_in[7];
  float* out = (float*)d_out;

  char* ws = (char*)d_ws;
  const size_t qbytes = (size_t)B_ * CQ_ * HW_ * sizeof(__bf16);  // 8.4 MB
  const size_t vbytes = (size_t)B_ * C_ * HW_ * sizeof(__bf16);   // 67 MB
  __bf16* qb = (__bf16*)ws;  ws += qbytes;
  __bf16* kb = (__bf16*)ws;  ws += qbytes;
  __bf16* vb = (__bf16*)ws;  ws += vbytes;
  float* mArr = (float*)ws;  ws += (size_t)B_ * HW_ * sizeof(float);
  float* sArr = (float*)ws;

  const dim3 blk(256);
  proj_kernel<<<dim3(B_, CQ_ / 16, HW_ / 128), blk, 0, stream>>>(x, Wq, bq, qb, CQ_);
  proj_kernel<<<dim3(B_, CQ_ / 16, HW_ / 128), blk, 0, stream>>>(x, Wk, bk, kb, CQ_);
  proj_kernel<<<dim3(B_, C_ / 16, HW_ / 128), blk, 0, stream>>>(x, Wv, bv, vb, C_);
  colstats_kernel<<<dim3(B_, W_), blk, 0, stream>>>(qb, kb, mArr, sArr);
  rowstats_kernel<<<dim3(B_, H_), blk, 0, stream>>>(qb, kb, mArr, sArr);
  colout_kernel<<<dim3(B_, W_), blk, 0, stream>>>(qb, kb, vb, mArr, sArr, out);
  rowout_kernel<<<dim3(B_, H_), blk, 0, stream>>>(qb, kb, vb, mArr, sArr, x, gamma, out);
}
